// AdditiveAttention_14817637171506
// MI455X (gfx1250) — compile-verified, matrix operand order checked by reading, not run on hardware
//
#include <hip/hip_runtime.h>
#include <math.h>

// Problem sizes (fixed by reference)
#define B_   4
#define TQ_  256
#define TK_  512
#define DQ_  512
#define DV_  512
#define U_   256
#define TROWS 4   // query rows per score workgroup (k-row L2 reuse)

typedef __attribute__((ext_vector_type(16))) __bf16 v16bf;
typedef __attribute__((ext_vector_type(8)))  float  v8f;

// ---------------- f32 -> bf16 elementwise convert ----------------
__global__ void cvt_bf16_kernel(const float* __restrict__ in,
                                __bf16* __restrict__ out, int n) {
  int i = blockIdx.x * blockDim.x + threadIdx.x;
  if (i < n) out[i] = (__bf16)in[i];
}

// ---------------- f32 [batch,R,C] -> bf16 [batch,C,R] -------------
__global__ void transpose_bf16_kernel(const float* __restrict__ in,
                                      __bf16* __restrict__ out,
                                      int R, int C, int n) {
  int i = blockIdx.x * blockDim.x + threadIdx.x;
  if (i >= n) return;
  int rc  = R * C;
  int b   = i / rc;
  int rem = i - b * rc;
  int r   = rem / C;
  int c   = rem - r * C;
  out[(size_t)b * rc + (size_t)c * R + r] = (__bf16)in[i];
}

// Load one 16x32 bf16 operand fragment (ISA 7.12.2 16-bit A layout):
// lanes 0-15 -> row m0+lane, K = k0+0..7 and k0+16..23
// lanes16-31 -> row m0+lane-16, K = k0+8..15 and k0+24..31
__device__ __forceinline__ v16bf load_frag16(const __bf16* __restrict__ base,
                                             int ld, int row0, int k0, int lane) {
  int r  = row0 + (lane & 15);
  int kh = (lane >> 4) << 3;              // 0 or 8
  const __bf16* p = base + (size_t)r * ld + k0 + kh;
  union { uint4 u[2]; v16bf v; } u;
  u.u[0] = *(const uint4*)(p);            // K = k0+kh .. +7   -> elems 0..7
  u.u[1] = *(const uint4*)(p + 16);       // K = k0+kh+16 .. +23 -> elems 8..15
  return u.v;
}

// C[M,N](f32) = A[M,K](bf16, row-major) x Bt[N,K](bf16, row-major = B^T)
// One wave (32 threads) per 16x16 output tile; batched via blockIdx.z.
__global__ void __launch_bounds__(32)
gemm_bf16_wmma(const __bf16* __restrict__ A,  int lda, long sA,
               const __bf16* __restrict__ Bt, int ldb, long sB,
               float* __restrict__ C,         int ldc, long sC, int K) {
  int lane = threadIdx.x & 31;
  int n0 = blockIdx.x * 16;
  int m0 = blockIdx.y * 16;
  int z  = blockIdx.z;
  const __bf16* Ab = A  + (size_t)z * sA;
  const __bf16* Bb = Bt + (size_t)z * sB;
  float*        Cb = C  + (size_t)z * sC;

  v8f acc = {};
  for (int k = 0; k < K; k += 32) {
    v16bf a = load_frag16(Ab, lda, m0, k, lane);
    v16bf b = load_frag16(Bb, ldb, n0, k, lane);
    acc = __builtin_amdgcn_wmma_f32_16x16x32_bf16(
        /*neg_a=*/false, a, /*neg_b=*/false, b,
        /*c_mod=*/(short)0, acc, /*reuse_a=*/false, /*reuse_b=*/false);
  }
  // C/D layout: acc[j]: lanes0-15 -> (M=m0+j, N=n0+lane); lanes16-31 -> (M=m0+j+8, N=n0+lane-16)
  int n  = n0 + (lane & 15);
  int mb = m0 + ((lane >> 4) << 3);
#pragma unroll
  for (int j = 0; j < 8; ++j)
    Cb[(size_t)(mb + j) * ldc + n] = acc[j];
}

// ---------------- fused tanh-score + softmax ----------------------
// One 256-thread block handles TROWS query rows of one batch.
// Lane l of each wave owns u = l, l+32, ..., l+224 (coalesced k loads);
// each wave produces one s per iteration (8 s in flight per block).
__global__ void __launch_bounds__(256)
score_softmax_kernel(const float* __restrict__ qp,   // [B,TQ,U]
                     const float* __restrict__ kp,   // [B,TK,U]
                     const float* __restrict__ scale,// [U]
                     float* __restrict__ attn_f,     // [B,TQ,TK] (d_out)
                     __bf16* __restrict__ attn_b) {  // [B,TQ,TK] (ws, for WMMA)
  __shared__ float sc[TROWS][TK_];
  __shared__ float red[8];
  int tid  = threadIdx.x;
  int lane = tid & 31, w = tid >> 5;
  int blk  = blockIdx.x;                  // B*TQ/TROWS blocks
  int b    = blk / (TQ_ / TROWS);
  int t0   = (blk % (TQ_ / TROWS)) * TROWS;

  float qv[TROWS][8], sv[8];
#pragma unroll
  for (int j = 0; j < 8; ++j) sv[j] = scale[lane + 32 * j];
#pragma unroll
  for (int r = 0; r < TROWS; ++r) {
    const float* qrow = qp + (size_t)(b * TQ_ + t0 + r) * U_;
#pragma unroll
    for (int j = 0; j < 8; ++j) qv[r][j] = qrow[lane + 32 * j];
  }

  const float* kb = kp + (size_t)b * TK_ * U_;
  for (int s = w; s < TK_; s += 8) {
    const float* krow = kb + (size_t)s * U_;
    float acc[TROWS];
#pragma unroll
    for (int r = 0; r < TROWS; ++r) acc[r] = 0.f;
#pragma unroll
    for (int j = 0; j < 8; ++j) {
      float kvv = krow[lane + 32 * j];    // coalesced 128B per wave
#pragma unroll
      for (int r = 0; r < TROWS; ++r) {
        // tanh(x) = 1 - 2/(exp(2x)+1): one v_exp_f32 + one v_rcp_f32
        float e  = __expf(2.0f * (qv[r][j] + kvv));
        float th = 1.0f - 2.0f * __builtin_amdgcn_rcpf(e + 1.0f);
        acc[r] = fmaf(sv[j], th, acc[r]);
      }
    }
#pragma unroll
    for (int r = 0; r < TROWS; ++r) {
      float a = acc[r];
#pragma unroll
      for (int off = 16; off > 0; off >>= 1) a += __shfl_xor(a, off, 32);
      if (lane == 0) sc[r][s] = a;
    }
  }
  __syncthreads();

  // softmax over TK=512 per row: each thread owns 2 entries
  for (int r = 0; r < TROWS; ++r) {
    float x0 = sc[r][tid], x1 = sc[r][tid + 256];
    float lm = fmaxf(x0, x1);
#pragma unroll
    for (int off = 16; off > 0; off >>= 1) lm = fmaxf(lm, __shfl_xor(lm, off, 32));
    if (lane == 0) red[w] = lm;
    __syncthreads();
    float gm = red[0];
#pragma unroll
    for (int i = 1; i < 8; ++i) gm = fmaxf(gm, red[i]);
    float e0 = __expf(x0 - gm), e1 = __expf(x1 - gm);
    float ls = e0 + e1;
    __syncthreads();                       // red reads done before rewrite
#pragma unroll
    for (int off = 16; off > 0; off >>= 1) ls += __shfl_xor(ls, off, 32);
    if (lane == 0) red[w] = ls;
    __syncthreads();
    float gs = red[0];
#pragma unroll
    for (int i = 1; i < 8; ++i) gs += red[i];
    float inv = __builtin_amdgcn_rcpf(gs);
    size_t orow = (size_t)(b * TQ_ + t0 + r) * TK_;
    float a0 = e0 * inv, a1 = e1 * inv;
    attn_f[orow + tid]       = a0;
    attn_f[orow + tid + 256] = a1;
    attn_b[orow + tid]       = (__bf16)a0;
    attn_b[orow + tid + 256] = (__bf16)a1;
    __syncthreads();                       // before next row reuses red
  }
}

// ------------------------------------------------------------------
extern "C" void kernel_launch(void* const* d_in, const int* in_sizes, int n_in,
                              void* d_out, int out_size, void* d_ws, size_t ws_size,
                              hipStream_t stream) {
  const float* query = (const float*)d_in[0]; // [B,TQ,DQ]
  const float* value = (const float*)d_in[1]; // [B,TK,DV]
  const float* W1    = (const float*)d_in[2]; // [DQ,U]
  const float* W2    = (const float*)d_in[3]; // [DV,U]
  const float* scale = (const float*)d_in[4]; // [U]

  float* context = (float*)d_out;                               // [B,TQ,DV]
  float* attn    = (float*)d_out + (size_t)B_ * TQ_ * DV_;      // [B,TQ,TK]

  char* p = (char*)d_ws;
  __bf16* query_bf  = (__bf16*)p;  p += (size_t)B_ * TQ_ * DQ_ * 2;
  __bf16* value_bf  = (__bf16*)p;  p += (size_t)B_ * TK_ * DV_ * 2;
  __bf16* valueT_bf = (__bf16*)p;  p += (size_t)B_ * TK_ * DV_ * 2;
  __bf16* W1T       = (__bf16*)p;  p += (size_t)DQ_ * U_ * 2;
  __bf16* W2T       = (__bf16*)p;  p += (size_t)DV_ * U_ * 2;
  float*  qproj     = (float*)p;   p += (size_t)B_ * TQ_ * U_ * 4;
  float*  kproj     = (float*)p;   p += (size_t)B_ * TK_ * U_ * 4;
  __bf16* attn_bf   = (__bf16*)p;  p += (size_t)B_ * TQ_ * TK_ * 2;

  int n;
  n = B_ * TQ_ * DQ_;
  cvt_bf16_kernel<<<(n + 255) / 256, 256, 0, stream>>>(query, query_bf, n);
  n = B_ * TK_ * DV_;
  cvt_bf16_kernel<<<(n + 255) / 256, 256, 0, stream>>>(value, value_bf, n);
  n = B_ * TK_ * DV_;
  transpose_bf16_kernel<<<(n + 255) / 256, 256, 0, stream>>>(value, valueT_bf, TK_, DV_, n);
  n = DQ_ * U_;
  transpose_bf16_kernel<<<(n + 255) / 256, 256, 0, stream>>>(W1, W1T, DQ_, U_, n);
  n = DV_ * U_;
  transpose_bf16_kernel<<<(n + 255) / 256, 256, 0, stream>>>(W2, W2T, DV_, U_, n);

  // qproj[1024,256] = query_bf[1024,512] x W1 (via W1T[256,512])
  gemm_bf16_wmma<<<dim3(U_ / 16, (B_ * TQ_) / 16, 1), 32, 0, stream>>>(
      query_bf, DQ_, 0, W1T, DQ_, 0, qproj, U_, 0, DQ_);
  // kproj[2048,256] = value_bf[2048,512] x W2 (via W2T[256,512])
  gemm_bf16_wmma<<<dim3(U_ / 16, (B_ * TK_) / 16, 1), 32, 0, stream>>>(
      value_bf, DV_, 0, W2T, DV_, 0, kproj, U_, 0, DV_);

  // fused tanh-score + softmax
  score_softmax_kernel<<<(B_ * TQ_) / TROWS, 256, 0, stream>>>(
      qproj, kproj, scale, attn, attn_bf);

  // context[b][256,512] = attn_bf[b][256,512] x value[b] (via valueT_bf[b][512,512])
  gemm_bf16_wmma<<<dim3(DV_ / 16, TQ_ / 16, B_), 32, 0, stream>>>(
      attn_bf, TK_, (long)TQ_ * TK_,
      valueT_bf, TK_, (long)DV_ * TK_,
      context, DV_, (long)TQ_ * DV_, TK_);
}
